// MultiHeadAttention_12756052869531
// MI455X (gfx1250) — matrix operand order repaired and checked by reading, not compiled, not run on hardware
//
#include <hip/hip_runtime.h>
#include <hip/hip_bf16.h>

typedef __bf16 bf16;
typedef bf16  v16bf __attribute__((ext_vector_type(16)));
typedef float v8f   __attribute__((ext_vector_type(8)));
typedef float v4f   __attribute__((ext_vector_type(4)));
typedef unsigned int v4u __attribute__((ext_vector_type(4)));
typedef unsigned int v8u __attribute__((ext_vector_type(8)));
typedef int v4i __attribute__((ext_vector_type(4)));

#define B_     4
#define S_     1024
#define D_     1024
#define H_     16
#define DEPTH_ 64
#define NROW   (B_ * S_)    // 4096
#define SROWPAD 40          // LDS row stride in bf16 (80B, 16B aligned, bank-spread)

#if __has_builtin(__builtin_amdgcn_global_load_async_to_lds_b128) && \
    __has_builtin(__builtin_amdgcn_s_wait_asynccnt)
#define HAVE_ASYNC_LDS 1
#else
#define HAVE_ASYNC_LDS 0
#endif

// ---------- scalar fp32 -> bf16 (RNE) ----------
__device__ __forceinline__ unsigned f2bfraw(float f) {
  union { float f; unsigned u; } in; in.f = f;
  unsigned u = in.u;
  return (u + 0x7FFFu + ((u >> 16) & 1u)) >> 16;
}
__device__ __forceinline__ bf16 f2bf(float f) {
  union { unsigned short s; bf16 b; } out;
  out.s = (unsigned short)f2bfraw(f);
  return out.b;
}

// ---------- async global -> LDS copy (16B per thread) ----------
__device__ __forceinline__ void cp16_g2s(const bf16* g, bf16* l) {
#if HAVE_ASYNC_LDS
  __builtin_amdgcn_global_load_async_to_lds_b128(
      (__attribute__((address_space(1))) v4i*)g,
      (__attribute__((address_space(3))) v4i*)l, 0, 0);
#else
  *(v4u*)l = *(const v4u*)g;
#endif
}
__device__ __forceinline__ void wait_async() {
#if HAVE_ASYNC_LDS
  __builtin_amdgcn_s_wait_asynccnt(0);
#endif
}

// ---------- fragment loaders ----------
// A operand 16x32 (MxK): lane m=lane&15, hf=lane>>4
__device__ __forceinline__ v16bf ldA_bf(const bf16* p, int ld, int rowBase, int kBase, int lane) {
  int m = lane & 15, hf = lane >> 4;
  const bf16* r = p + (size_t)(rowBase + m) * ld + kBase + hf * 8;
  v4u lo = *(const v4u*)r;
  v4u hi = *(const v4u*)(r + 16);
  v8u c;
#pragma unroll
  for (int i = 0; i < 4; ++i) { c[i] = lo[i]; c[4 + i] = hi[i]; }
  return __builtin_bit_cast(v16bf, c);
}

__device__ __forceinline__ v16bf ldA_f32cvt(const float* p, int ld, int rowBase, int kBase, int lane) {
  int m = lane & 15, hf = lane >> 4;
  const float* r = p + (size_t)(rowBase + m) * ld + kBase + hf * 8;
  v4f f0 = *(const v4f*)r;
  v4f f1 = *(const v4f*)(r + 4);
  v4f f2 = *(const v4f*)(r + 16);
  v4f f3 = *(const v4f*)(r + 20);
  v8u c;
#pragma unroll
  for (int i = 0; i < 2; ++i) {
    c[i]     = f2bfraw(f0[2*i]) | (f2bfraw(f0[2*i+1]) << 16);
    c[2 + i] = f2bfraw(f1[2*i]) | (f2bfraw(f1[2*i+1]) << 16);
    c[4 + i] = f2bfraw(f2[2*i]) | (f2bfraw(f2[2*i+1]) << 16);
    c[6 + i] = f2bfraw(f3[2*i]) | (f2bfraw(f3[2*i+1]) << 16);
  }
  return __builtin_bit_cast(v16bf, c);
}

// B operand 32x16 (KxN) from global: lane n=lane&15 reads k 8hf..8hf+7 and 16+8hf..16+8hf+7 of row n
__device__ __forceinline__ v16bf ldB_bf(const bf16* p, int ld, int colBase, int kBase, int lane) {
  int n = lane & 15, hf = lane >> 4;
  const bf16* r = p + (size_t)(colBase + n) * ld + kBase + hf * 8;
  v4u lo = *(const v4u*)r;
  v4u hi = *(const v4u*)(r + 16);
  v8u c;
#pragma unroll
  for (int i = 0; i < 4; ++i) { c[i] = lo[i]; c[4 + i] = hi[i]; }
  return __builtin_bit_cast(v16bf, c);
}

// B operand from LDS-staged tile: 64 rows x 32 k, row stride SROWPAD
__device__ __forceinline__ v16bf ldB_lds(const bf16* s, int colOff, int lane) {
  int n = lane & 15, hf = lane >> 4;
  const bf16* r = s + (size_t)(colOff + n) * SROWPAD + hf * 8;
  v4u lo = *(const v4u*)r;
  v4u hi = *(const v4u*)(r + 16);
  v8u c;
#pragma unroll
  for (int i = 0; i < 4; ++i) { c[i] = lo[i]; c[4 + i] = hi[i]; }
  return __builtin_bit_cast(v16bf, c);
}

__device__ __forceinline__ v8f wmma_bf16(v16bf a, v16bf b, v8f c) {
  return __builtin_amdgcn_wmma_f32_16x16x32_bf16(false, a, false, b, (short)0, c, false, false);
}

// ---------- kernel 0: one-shot fp32 -> bf16 conversion ----------
__global__ void k_convert(const float* __restrict__ q, const float* __restrict__ k,
                          const float* __restrict__ v,
                          const float* __restrict__ Wq, const float* __restrict__ Wk,
                          const float* __restrict__ Wv, const float* __restrict__ Wo,
                          bf16* __restrict__ Xbf, bf16* __restrict__ Wbf) {
  int z = blockIdx.z;
  const float* src; bf16* dst; size_t n;
  if (z < 3) {
    src = z == 0 ? q : (z == 1 ? k : v);
    dst = Xbf + (size_t)z * NROW * D_;
    n = (size_t)NROW * D_;
  } else {
    src = z == 3 ? Wq : (z == 4 ? Wk : (z == 5 ? Wv : Wo));
    dst = Wbf + (size_t)(z - 3) * D_ * D_;
    n = (size_t)D_ * D_;
  }
  size_t idx = ((size_t)blockIdx.x * 256 + threadIdx.x) * 8;
  if (idx >= n) return;
  v4f f0 = *(const v4f*)(src + idx);
  v4f f1 = *(const v4f*)(src + idx + 4);
  v4u o;
  o[0] = f2bfraw(f0[0]) | (f2bfraw(f0[1]) << 16);
  o[1] = f2bfraw(f0[2]) | (f2bfraw(f0[3]) << 16);
  o[2] = f2bfraw(f1[0]) | (f2bfraw(f1[1]) << 16);
  o[3] = f2bfraw(f1[2]) | (f2bfraw(f1[3]) << 16);
  *(v4u*)(dst + idx) = o;
}

// ---------- kernel 1: fused QKV projections  Y = X @ W^T + b ----------
__global__ void k_proj_qkv(const bf16* __restrict__ Xbf, const bf16* __restrict__ Wbf,
                           const float* __restrict__ bq, const float* __restrict__ bk,
                           const float* __restrict__ bv,
                           bf16* __restrict__ Qbf, bf16* __restrict__ Kbf,
                           bf16* __restrict__ Vt) {
  __shared__ bf16 sB[2][64][SROWPAD];
  int tid = threadIdx.x;
  int lane = tid & 31, wave = tid >> 5;
  int rowTile = blockIdx.x * 8 + wave;        // 0..255
  int colG    = blockIdx.y;                   // 0..15
  int which   = blockIdx.z;                   // 0=q 1=k 2=v
  const bf16* X = Xbf + (size_t)which * NROW * D_;
  const bf16* W = Wbf + (size_t)which * D_ * D_;
  const float* bias = which == 0 ? bq : (which == 1 ? bk : bv);

  int rowBase = rowTile * 16, colBase = colG * 64;

  int srow = tid >> 2, sseg = tid & 3;
  const bf16* gsrc = W + (size_t)(colBase + srow) * D_ + sseg * 8;
  bf16* ldst0 = &sB[0][srow][sseg * 8];
  bf16* ldst1 = &sB[1][srow][sseg * 8];

  cp16_g2s(gsrc, ldst0);

  v8f acc[4] = {};
  for (int it = 0; it < D_ / 32; ++it) {
    int kb = it * 32;
    wait_async();
    __syncthreads();
    if (kb + 32 < D_) cp16_g2s(gsrc + kb + 32, (it & 1) ? ldst0 : ldst1);
    const bf16* sbuf = &sB[it & 1][0][0];
    v16bf a  = ldA_bf(X, D_, rowBase, kb, lane);
    v16bf b0 = ldB_lds(sbuf, 0,  lane);
    v16bf b1 = ldB_lds(sbuf, 16, lane);
    v16bf b2 = ldB_lds(sbuf, 32, lane);
    v16bf b3 = ldB_lds(sbuf, 48, lane);
    acc[0] = wmma_bf16(a, b0, acc[0]);
    acc[1] = wmma_bf16(a, b1, acc[1]);
    acc[2] = wmma_bf16(a, b2, acc[2]);
    acc[3] = wmma_bf16(a, b3, acc[3]);
  }

  int n = lane & 15, hf = lane >> 4;
#pragma unroll
  for (int t = 0; t < 4; ++t) {
    int col = colBase + t * 16 + n;
    int h = col >> 6, d = col & 63;
    float bia = bias[col];
#pragma unroll
    for (int j = 0; j < 8; ++j) {
      int row = rowBase + hf * 8 + j;
      int bb = row >> 10, s = row & 1023;
      bf16 val = f2bf(acc[t][j] + bia);
      if (which == 2) {
        Vt[((size_t)(bb * H_ + h) * DEPTH_ + d) * S_ + s] = val;
      } else {
        bf16* dst = (which == 0) ? Qbf : Kbf;
        dst[((size_t)(bb * H_ + h) * S_ + s) * DEPTH_ + d] = val;
      }
    }
  }
}

// ---------- kernel 2: logits = (Q K^T) * scale + mask * -1e9 ----------
__global__ void k_logits(const bf16* __restrict__ Qbf, const bf16* __restrict__ Kbf,
                         const float* __restrict__ mask, float* __restrict__ attn) {
  int lane = threadIdx.x & 31, wave = threadIdx.x >> 5;
  int qTile = blockIdx.x * 8 + wave;          // 0..63
  int kG    = blockIdx.y;                     // 0..15
  int bh    = blockIdx.z;                     // 0..63
  const bf16* Q = Qbf + (size_t)bh * S_ * DEPTH_;
  const bf16* K = Kbf + (size_t)bh * S_ * DEPTH_;
  int rowBase = qTile * 16, colBase = kG * 64;

  v8f acc[4] = {};
#pragma unroll
  for (int kb = 0; kb < DEPTH_; kb += 32) {
    v16bf a  = ldA_bf(Q, DEPTH_, rowBase, kb, lane);
    v16bf b0 = ldB_bf(K, DEPTH_, colBase + 0,  kb, lane);
    v16bf b1 = ldB_bf(K, DEPTH_, colBase + 16, kb, lane);
    v16bf b2 = ldB_bf(K, DEPTH_, colBase + 32, kb, lane);
    v16bf b3 = ldB_bf(K, DEPTH_, colBase + 48, kb, lane);
    acc[0] = wmma_bf16(a, b0, acc[0]);
    acc[1] = wmma_bf16(a, b1, acc[1]);
    acc[2] = wmma_bf16(a, b2, acc[2]);
    acc[3] = wmma_bf16(a, b3, acc[3]);
  }

  int n = lane & 15, hf = lane >> 4;
  int bb = bh >> 4;
  const float scale = 0.125f;                 // 1/sqrt(64)
  float* out = attn + (size_t)bh * S_ * S_;
#pragma unroll
  for (int t = 0; t < 4; ++t) {
    int col = colBase + t * 16 + n;
    float mk = mask[bb * S_ + col] * -1e9f;
#pragma unroll
    for (int j = 0; j < 8; ++j) {
      int row = rowBase + hf * 8 + j;
      out[(size_t)row * S_ + col] = acc[t][j] * scale + mk;
    }
  }
}

// ---------- kernel 3: row softmax, in place, one wave per row ----------
__global__ void k_softmax(float* __restrict__ attn) {
  int lane = threadIdx.x & 31, wave = threadIdx.x >> 5;
  size_t row = (size_t)blockIdx.x * 8 + wave;   // 65536 rows
  float* p = attn + row * S_;
  float r[32];
  float mx = -3.4e38f;
#pragma unroll
  for (int i = 0; i < 32; ++i) { r[i] = p[lane + i * 32]; mx = fmaxf(mx, r[i]); }
#pragma unroll
  for (int off = 16; off > 0; off >>= 1) mx = fmaxf(mx, __shfl_xor(mx, off, 32));
  float sum = 0.f;
#pragma unroll
  for (int i = 0; i < 32; ++i) { r[i] = __expf(r[i] - mx); sum += r[i]; }
#pragma unroll
  for (int off = 16; off > 0; off >>= 1) sum += __shfl_xor(sum, off, 32);
  float inv = 1.0f / sum;
#pragma unroll
  for (int i = 0; i < 32; ++i) p[lane + i * 32] = r[i] * inv;
}

// ---------- kernel 4: ctx = attn @ V (V pre-transposed [bh][d][s]) ----------
__global__ void k_ctx(const float* __restrict__ attn, const bf16* __restrict__ Vt,
                      bf16* __restrict__ ctx) {
  __shared__ bf16 sB[2][64][SROWPAD];
  int tid = threadIdx.x;
  int lane = tid & 31, wave = tid >> 5;
  int qTile = blockIdx.x * 8 + wave;          // 0..63
  int bh    = blockIdx.z;                     // 0..63
  const float* A  = attn + (size_t)bh * S_ * S_;
  const bf16*  Bv = Vt   + (size_t)bh * DEPTH_ * S_;   // [d][s], ld = S
  int rowBase = qTile * 16;

  int srow = tid >> 2, sseg = tid & 3;        // srow = d index 0..63
  const bf16* gsrc = Bv + (size_t)srow * S_ + sseg * 8;
  bf16* ldst0 = &sB[0][srow][sseg * 8];
  bf16* ldst1 = &sB[1][srow][sseg * 8];

  cp16_g2s(gsrc, ldst0);

  v8f acc[4] = {};
  for (int it = 0; it < S_ / 32; ++it) {
    int kb = it * 32;
    wait_async();
    __syncthreads();
    if (kb + 32 < S_) {
      cp16_g2s(gsrc + kb + 32, (it & 1) ? ldst0 : ldst1);
      __builtin_prefetch(A + (size_t)(rowBase + (lane & 15)) * S_ + kb + 32, 0, 1);
    }
    const bf16* sbuf = &sB[it & 1][0][0];
    v16bf a  = ldA_f32cvt(A, S_, rowBase, kb, lane);
    v16bf b0 = ldB_lds(sbuf, 0,  lane);
    v16bf b1 = ldB_lds(sbuf, 16, lane);
    v16bf b2 = ldB_lds(sbuf, 32, lane);
    v16bf b3 = ldB_lds(sbuf, 48, lane);
    acc[0] = wmma_bf16(a, b0, acc[0]);
    acc[1] = wmma_bf16(a, b1, acc[1]);
    acc[2] = wmma_bf16(a, b2, acc[2]);
    acc[3] = wmma_bf16(a, b3, acc[3]);
  }

  int n = lane & 15, hf = lane >> 4;
  int bb = bh >> 4, h = bh & 15;
#pragma unroll
  for (int t = 0; t < 4; ++t) {
    int dcol = t * 16 + n;
#pragma unroll
    for (int j = 0; j < 8; ++j) {
      int srowO = rowBase + hf * 8 + j;
      ctx[((size_t)bb * S_ + srowO) * D_ + h * DEPTH_ + dcol] = f2bf(acc[t][j]);
    }
  }
}

// ---------- kernel 5: out = ctx @ Wo^T + bo ----------
__global__ void k_outproj(const bf16* __restrict__ ctx, const bf16* __restrict__ Wob,
                          const float* __restrict__ bo, float* __restrict__ out) {
  __shared__ bf16 sB[2][64][SROWPAD];
  int tid = threadIdx.x;
  int lane = tid & 31, wave = tid >> 5;
  int rowTile = blockIdx.x * 8 + wave;        // 0..255
  int colG    = blockIdx.y;                   // 0..15
  int rowBase = rowTile * 16, colBase = colG * 64;

  int srow = tid >> 2, sseg = tid & 3;
  const bf16* gsrc = Wob + (size_t)(colBase + srow) * D_ + sseg * 8;
  bf16* ldst0 = &sB[0][srow][sseg * 8];
  bf16* ldst1 = &sB[1][srow][sseg * 8];

  cp16_g2s(gsrc, ldst0);

  v8f acc[4] = {};
  for (int it = 0; it < D_ / 32; ++it) {
    int kb = it * 32;
    wait_async();
    __syncthreads();
    if (kb + 32 < D_) cp16_g2s(gsrc + kb + 32, (it & 1) ? ldst0 : ldst1);
    const bf16* sbuf = &sB[it & 1][0][0];
    v16bf a  = ldA_bf(ctx, D_, rowBase, kb, lane);
    v16bf b0 = ldB_lds(sbuf, 0,  lane);
    v16bf b1 = ldB_lds(sbuf, 16, lane);
    v16bf b2 = ldB_lds(sbuf, 32, lane);
    v16bf b3 = ldB_lds(sbuf, 48, lane);
    acc[0] = wmma_bf16(a, b0, acc[0]);
    acc[1] = wmma_bf16(a, b1, acc[1]);
    acc[2] = wmma_bf16(a, b2, acc[2]);
    acc[3] = wmma_bf16(a, b3, acc[3]);
  }

  int n = lane & 15, hf = lane >> 4;
#pragma unroll
  for (int t = 0; t < 4; ++t) {
    int col = colBase + t * 16 + n;
    float bia = bo[col];
#pragma unroll
    for (int j = 0; j < 8; ++j) {
      int row = rowBase + hf * 8 + j;
      out[(size_t)row * D_ + col] = acc[t][j] + bia;
    }
  }
}

// ---------- launch ----------
extern "C" void kernel_launch(void* const* d_in, const int* in_sizes, int n_in,
                              void* d_out, int out_size, void* d_ws, size_t ws_size,
                              hipStream_t stream) {
  const float* q    = (const float*)d_in[0];
  const float* k    = (const float*)d_in[1];
  const float* v    = (const float*)d_in[2];
  const float* mask = (const float*)d_in[3];
  const float* Wq   = (const float*)d_in[4];
  const float* bq   = (const float*)d_in[5];
  const float* Wk   = (const float*)d_in[6];
  const float* bk   = (const float*)d_in[7];
  const float* Wv   = (const float*)d_in[8];
  const float* bv   = (const float*)d_in[9];
  const float* Wo   = (const float*)d_in[10];
  const float* bo   = (const float*)d_in[11];

  float* out  = (float*)d_out;
  float* attn = out + (size_t)B_ * S_ * D_;          // attn region of d_out

  const size_t M4 = (size_t)B_ * S_ * D_;            // 4M elements
  bf16* Qbf = (bf16*)d_ws;                            // [0,  4M)
  bf16* Kbf = Qbf + M4;                               // [4M, 8M)
  bf16* Vt  = Kbf + M4;                               // [8M, 12M)
  bf16* ctx = Vt  + M4;                               // [12M,16M)
  bf16* Xbf = ctx + M4;                               // [16M,28M)  3 x 4M
  bf16* Wbf = Xbf + 3 * M4;                           // [28M,32M)  4 x 1M

  dim3 blk(256);
  k_convert <<<dim3(2048, 1, 7), blk, 0, stream>>>(q, k, v, Wq, Wk, Wv, Wo, Xbf, Wbf);
  k_proj_qkv<<<dim3(32, 16, 3),  blk, 0, stream>>>(Xbf, Wbf, bq, bk, bv, Qbf, Kbf, Vt);
  k_logits  <<<dim3(8, 16, 64),  blk, 0, stream>>>(Qbf, Kbf, mask, attn);
  k_softmax <<<dim3(8192),       blk, 0, stream>>>(attn);
  k_ctx     <<<dim3(8, 1, 64),   blk, 0, stream>>>(attn, Vt, ctx);
  k_outproj <<<dim3(32, 16),     blk, 0, stream>>>(ctx, Wbf + (size_t)3 * D_ * D_, bo, out);
}
